// ConvLSTM_46626164965564
// MI455X (gfx1250) — matrix operand order repaired and checked by reading, not compiled, not run on hardware
//
#include <hip/hip_runtime.h>
#include <math.h>

// ---------------- types for WMMA fragments ----------------
typedef __attribute__((ext_vector_type(16))) __bf16 v16bf;
typedef __attribute__((ext_vector_type(8)))  float  v8f;

union AFrag { v16bf v; unsigned int d[8]; };
union BFrag { v16bf v; uint4 q[2]; };
union CFrag { v8f  v; float4 q[2]; float f[8]; };

// Problem constants
#define Bn   8
#define Tn   16
#define CIN  16
#define Hn   64
#define Wn   64
#define HID  64
#define CINH 80              // CIN + HID
#define CPAD 96              // CINH padded to multiple of 32
#define KTOT (9 * CPAD)      // 864
#define KCH  (KTOT / 32)     // 27 k-chunks
#define NCH  256             // f,i,g,o each 64
#define PIX  (Bn * Hn * Wn)  // 32768 pixels per timestep
#define MT   32              // pixels per block (2 M-tiles of 16)
#define STATE_ELEMS (Bn * HID * Hn * Wn)  // 2097152

__device__ __forceinline__ unsigned short f32_to_bf16(float f) {
    unsigned int u = __float_as_uint(f);
    u += 0x7fffu + ((u >> 16) & 1u);   // round-to-nearest-even
    return (unsigned short)(u >> 16);
}

__device__ __forceinline__ float sigmoidf_(float x) {
    return 1.0f / (1.0f + __expf(-x));
}

// ---------------- weight repack: bf16, B-fragment-major ----------------
// Wpack[((kc*16 + nt)*32 + lane)*16 + e] = W[K][N]
//   K = kc*32 + (lane>>4)*16 + e   (K = tap*96 + cpad; cpad<80 valid)
//   N = nt*16 + (lane&15)          (N: 0..63 f, 64..127 i, 128..191 g, 192..255 o)
__global__ __launch_bounds__(256) void pack_kernel(
    const float* __restrict__ Wf, const float* __restrict__ Wi,
    const float* __restrict__ Wc, const float* __restrict__ Wo,
    unsigned short* __restrict__ Wpack)
{
    int tid = blockIdx.x * 256 + threadIdx.x;          // 0 .. 221183
    int e    =  tid        & 15;
    int lane = (tid >> 4)  & 31;
    int nt   = (tid >> 9)  & 15;
    int kc   =  tid >> 13;                             // 0..26
    int K = kc * 32 + (lane >> 4) * 16 + e;
    int N = nt * 16 + (lane & 15);
    int tap = K / CPAD;
    int cp  = K - tap * CPAD;
    int dy = tap / 3, dx = tap - dy * 3;
    int gate = N >> 6;
    int oc   = N & 63;
    float v = 0.0f;
    if (cp < CINH) {
        int widx = ((oc * CINH + cp) * 3 + dy) * 3 + dx;
        if      (gate == 0) v = Wf[widx];
        else if (gate == 1) v = Wi[widx];
        else if (gate == 2) v = Wc[widx];
        else if (tap == 4)  v = Wo[oc * CINH + cp];    // linear = center tap only
    }
    Wpack[tid] = f32_to_bf16(v);
}

// ---------------- implicit-GEMM gate kernel ----------------
// Block = 128 threads (4 waves). Block handles 32 pixels (b, y, x0..x0+31)
// x all 256 gate channels. Wave: 2 M-tiles x 4 N-tiles, double-buffered B.
__global__ __launch_bounds__(128) void gates_kernel(
    const float* __restrict__ x, int t,
    const float* __restrict__ h,
    const unsigned short* __restrict__ Wpack,
    float* __restrict__ G)
{
    int blk = blockIdx.x;                 // 1024 = 8 * 64 * 2
    int xt  =  blk       & 1;
    int y   = (blk >> 1) & 63;
    int b   =  blk >> 7;
    int x0  = xt * MT;
    int tid = threadIdx.x;

    __shared__ __align__(16) unsigned short A[MT * KTOT];   // 55296 bytes

    // ---- stage A tile (32 pixels x 864 K) into LDS as bf16, zero 'SAME' halo
    for (int it = 0; it < (MT * KTOT) / 128; ++it) {        // 216 iters
        int idx = it * 128 + tid;
        int m   = idx / KTOT;
        int k   = idx - m * KTOT;
        int tap = k / CPAD;
        int cp  = k - tap * CPAD;
        int dy = tap / 3, dx = tap - dy * 3;
        int yy = y + dy - 1;
        int xx = x0 + m + dx - 1;
        float v = 0.0f;
        if (cp < CINH && (unsigned)yy < (unsigned)Hn && (unsigned)xx < (unsigned)Wn) {
            if (cp < CIN)
                v = x[(((b * Tn + t) * CIN + cp) * Hn + yy) * Wn + xx];
            else
                v = h[((b * HID + (cp - CIN)) * Hn + yy) * Wn + xx];
        }
        A[idx] = f32_to_bf16(v);
    }
    __syncthreads();

    const unsigned int* Ad = (const unsigned int*)A;
    int lane = tid & 31;
    int wave = tid >> 5;
    int half = lane >> 4;
    int mn   = lane & 15;
    int hoff = half * 8;

    auto loadA = [&](int m, int kc) {
        AFrag a;
        int abase = (m * KTOT + kc * 32) >> 1;   // dword index
#pragma unroll
        for (int j = 0; j < 4; ++j) a.d[j]     = Ad[abase + hoff + j];
#pragma unroll
        for (int j = 0; j < 4; ++j) a.d[4 + j] = Ad[abase + hoff + 4 + j];
        return a;
    };
    auto loadB = [&](BFrag* dst, int kc) {
#pragma unroll
        for (int i = 0; i < 4; ++i) {
            int nt = wave * 4 + i;
            const uint4* wp = (const uint4*)(Wpack +
                ((((kc * 16 + nt) * 32) + lane) << 4));
            dst[i].q[0] = wp[0];
            dst[i].q[1] = wp[1];
        }
    };

    CFrag acc[2][4];
#pragma unroll
    for (int mt = 0; mt < 2; ++mt)
#pragma unroll
        for (int i = 0; i < 4; ++i)
#pragma unroll
            for (int r = 0; r < 8; ++r) acc[mt][i].f[r] = 0.0f;

    BFrag bw[2][4];
    loadB(bw[0], 0);

#pragma unroll
    for (int kc = 0; kc < KCH; ++kc) {
        int cur = kc & 1;
        if (kc + 1 < KCH) loadB(bw[cur ^ 1], kc + 1);   // prefetch next chunk
        AFrag a0 = loadA(mn, kc);
        AFrag a1 = loadA(16 + mn, kc);
#pragma unroll
        for (int i = 0; i < 4; ++i) {
            acc[0][i].v = __builtin_amdgcn_wmma_f32_16x16x32_bf16(
                false, a0.v, false, bw[cur][i].v, (short)0, acc[0][i].v, false, false);
            acc[1][i].v = __builtin_amdgcn_wmma_f32_16x16x32_bf16(
                false, a1.v, false, bw[cur][i].v, (short)0, acc[1][i].v, false, false);
        }
    }

    // ---- store raw pre-activations: G[ch][pix]; lane owns 8 consecutive pixels
    int pixbase = ((b * Hn + y) * Wn) + x0;
#pragma unroll
    for (int mt = 0; mt < 2; ++mt) {
#pragma unroll
        for (int i = 0; i < 4; ++i) {
            int nt = wave * 4 + i;
            int ch = nt * 16 + mn;
            float4* dst = (float4*)(G + (size_t)ch * PIX + pixbase +
                                    mt * 16 + half * 8);
            dst[0] = acc[mt][i].q[0];
            dst[1] = acc[mt][i].q[1];
        }
    }
}

// ---------------- elementwise LSTM cell update ----------------
__global__ __launch_bounds__(256) void lstm_update_kernel(
    const float* __restrict__ G,
    const float* __restrict__ bf_, const float* __restrict__ bi_,
    const float* __restrict__ bc_, const float* __restrict__ bo_,
    float* __restrict__ cst, float* __restrict__ hst,
    float* __restrict__ out, int write_out)
{
    int tid = blockIdx.x * 256 + threadIdx.x;    // 0 .. STATE_ELEMS-1
    int b   = tid >> 18;                         // / (HID*H*W)
    int rem = tid & ((1 << 18) - 1);
    int j   = rem >> 12;                         // hidden channel
    int s   = rem & 4095;                        // spatial
    int pix = (b << 12) + s;

    float f = sigmoidf_(G[(size_t)(      j) * PIX + pix] + bf_[j]);
    float i = sigmoidf_(G[(size_t)( 64 + j) * PIX + pix] + bi_[j]);
    float g = tanhf    (G[(size_t)(128 + j) * PIX + pix] + bc_[j]);
    float o = sigmoidf_(G[(size_t)(192 + j) * PIX + pix] + bo_[j]);

    float cn = cst[tid] * f + i * g;
    float hn = tanhf(cn) * o;
    cst[tid] = cn;
    hst[tid] = hn;
    if (write_out) {
        out[tid]               = hn;   // hT first
        out[STATE_ELEMS + tid] = cn;   // then cT
    }
}

extern "C" void kernel_launch(void* const* d_in, const int* in_sizes, int n_in,
                              void* d_out, int out_size, void* d_ws, size_t ws_size,
                              hipStream_t stream) {
    (void)in_sizes; (void)n_in; (void)out_size; (void)ws_size;

    const float* x  = (const float*)d_in[0];
    const float* Wf = (const float*)d_in[1];
    const float* bf = (const float*)d_in[2];
    const float* Wi = (const float*)d_in[3];
    const float* bi = (const float*)d_in[4];
    const float* Wc = (const float*)d_in[5];
    const float* bc = (const float*)d_in[6];
    const float* Wo = (const float*)d_in[7];
    const float* bo = (const float*)d_in[8];
    float* out = (float*)d_out;

    char* ws = (char*)d_ws;
    const size_t stateBytes = (size_t)STATE_ELEMS * 4;            // 8 MB
    float*          h_st = (float*)(ws);
    float*          c_st = (float*)(ws + stateBytes);
    float*          G    = (float*)(ws + 2 * stateBytes);
    unsigned short* Wp   = (unsigned short*)(ws + 2 * stateBytes +
                                             (size_t)NCH * PIX * 4);

    // zero h and c states (contiguous)
    hipMemsetAsync(h_st, 0, 2 * stateBytes, stream);

    // repack weights to bf16 fragment order: 27*16*32*16 = 221184 threads
    pack_kernel<<<221184 / 256, 256, 0, stream>>>(Wf, Wi, Wc, Wo, Wp);

    for (int t = 0; t < Tn; ++t) {
        gates_kernel<<<Bn * Hn * (Wn / MT), 128, 0, stream>>>(x, t, h_st, Wp, G);
        lstm_update_kernel<<<STATE_ELEMS / 256, 256, 0, stream>>>(
            G, bf, bi, bc, bo, c_st, h_st, out, (t == Tn - 1) ? 1 : 0);
    }
}
